// Cross_MultiAttention_37933151158475
// MI455X (gfx1250) — compile-verified, matrix operand order checked by reading, not run on hardware
//
#include <hip/hip_runtime.h>
#include <hip/hip_bf16.h>
#include <stdint.h>

// ---------------- problem constants ----------------
#define BB      4
#define SQ      1024
#define SKV     4096
#define DIMQ    1024
#define DIMKV   768
#define NH      16
#define DH      64
#define DIMOUT  256

// padded LDS row strides (16B-aligned, spreads DS banks)
#define LDH     40   // f16 tile row stride (32 data + 8 pad halves) = 80B
#define LDF     36   // f32 tile row stride (32 data + 4 pad floats) = 144B

typedef __attribute__((ext_vector_type(16))) _Float16 v16h;
typedef __attribute__((ext_vector_type(8)))  _Float16 half8;
typedef __attribute__((ext_vector_type(8)))  float    v8f;
typedef __attribute__((ext_vector_type(4)))  float    f4;
typedef __attribute__((ext_vector_type(4)))  int      i4;

// ---------------------------------------------------------------------------
// Async global -> LDS copy (CDNA5 GLOBAL_LOAD_ASYNC_TO_LDS_B128, ASYNCcnt).
// Builtin signature (from hipcc diagnostic): param0 = v4i32 addrspace(1)*.
// Guarded: falls back to synchronous LDS staging if builtin unavailable.
// ---------------------------------------------------------------------------
#if defined(__has_builtin)
#if __has_builtin(__builtin_amdgcn_global_load_async_to_lds_b128)
#define HAVE_ASYNC_LDS 1
#endif
#if __has_builtin(__builtin_amdgcn_s_wait_asynccnt)
#define HAVE_WAIT_ASYNC 1
#endif
#endif

typedef __attribute__((address_space(1))) i4 gas_i4;
typedef __attribute__((address_space(3))) i4 las_i4;

__device__ __forceinline__ void cp16(void* dst_lds, const void* src_glob) {
#if defined(HAVE_ASYNC_LDS)
  __builtin_amdgcn_global_load_async_to_lds_b128(
      (gas_i4*)(unsigned long long)(uintptr_t)src_glob,
      (las_i4*)(unsigned int)(uintptr_t)dst_lds,
      0, 0);
#else
  *(i4*)dst_lds = *(const i4*)src_glob;
#endif
}

#if defined(HAVE_ASYNC_LDS)
#if defined(HAVE_WAIT_ASYNC)
#define WAIT_ASYNC(n) __builtin_amdgcn_s_wait_asynccnt(n)
#else
#define WAIT_ASYNC(n) asm volatile("s_wait_asynccnt %0" ::"n"(n) : "memory")
#endif
#else
#define WAIT_ASYNC(n) ((void)0)
#endif

// ---------------------------------------------------------------------------
// Stage one 64x32 A tile + 64x32 B tile into LDS. 128 threads; each thread
// copies 16B chunks. f16/f16 variant: 4 async ops/thread. f32-A variant: 6.
// ---------------------------------------------------------------------------
__device__ __forceinline__ void stage_f16f16(const _Float16* Ag, int lda,
                                             const _Float16* Bg, int ldb,
                                             int k0, _Float16* sA, _Float16* sB) {
  const int t = threadIdx.x;
#pragma unroll
  for (int u = 0; u < 2; ++u) {
    const int c = t + u * 128;          // 256 chunks per tile
    const int row = c >> 2;
    const int co  = (c & 3) << 3;       // half offset within 32-wide k chunk
    cp16(sA + row * LDH + co, Ag + (size_t)row * lda + k0 + co);
    cp16(sB + row * LDH + co, Bg + (size_t)row * ldb + k0 + co);
  }
}

__device__ __forceinline__ void stage_f32f16(const float* Ag, int lda,
                                             const _Float16* Bg, int ldb,
                                             int k0, float* sA, _Float16* sB) {
  const int t = threadIdx.x;
#pragma unroll
  for (int u = 0; u < 4; ++u) {         // A: 64 rows x 32 f32 = 512 chunks
    const int c = t + u * 128;
    const int row = c >> 3;
    const int co  = (c & 7) << 2;       // float offset
    cp16(sA + row * LDF + co, Ag + (size_t)row * lda + k0 + co);
  }
#pragma unroll
  for (int u = 0; u < 2; ++u) {
    const int c = t + u * 128;
    const int row = c >> 2;
    const int co  = (c & 3) << 3;
    cp16(sB + row * LDH + co, Bg + (size_t)row * ldb + k0 + co);
  }
}

// ---------------------------------------------------------------------------
// WMMA fragment build from an LDS tile (row0 pre-applied by caller).
// 16-bit A/B layout (ISA 7.12.2): lane r=lane&15 -> row/col r, hi half of the
// wave takes K+8 / K+24 chunks.
// ---------------------------------------------------------------------------
__device__ __forceinline__ v16h frag_lds_f16(const _Float16* tile) {
  const int lane = threadIdx.x & 31;
  const int r    = lane & 15;
  const int hi   = (lane >> 4) << 3;
  const _Float16* p = tile + r * LDH;
  half8 c0 = *(const half8*)(p + hi);
  half8 c1 = *(const half8*)(p + 16 + hi);
  v16h f;
#pragma unroll
  for (int i = 0; i < 8; ++i) { f[i] = c0[i]; f[i + 8] = c1[i]; }
  return f;
}

__device__ __forceinline__ v16h frag_lds_f32(const float* tile) {
  const int lane = threadIdx.x & 31;
  const int r    = lane & 15;
  const int hi   = (lane >> 4) << 3;
  const float* p = tile + r * LDF;
  f4 a0 = *(const f4*)(p + hi);
  f4 a1 = *(const f4*)(p + hi + 4);
  f4 b0 = *(const f4*)(p + 16 + hi);
  f4 b1 = *(const f4*)(p + 20 + hi);
  v16h f;
#pragma unroll
  for (int i = 0; i < 4; ++i) {
    f[i]      = (_Float16)a0[i];
    f[i + 4]  = (_Float16)a1[i];
    f[i + 8]  = (_Float16)b0[i];
    f[i + 12] = (_Float16)b1[i];
  }
  return f;
}

// ---------------------------------------------------------------------------
// f32 -> f16 elementwise conversion, grid-stride.
// ---------------------------------------------------------------------------
__global__ void cvt_f32_f16_kernel(const float* __restrict__ src,
                                   _Float16* __restrict__ dst, long n) {
  long i = (long)blockIdx.x * blockDim.x + threadIdx.x;
  long stride = (long)gridDim.x * blockDim.x;
  for (; i < n; i += stride) dst[i] = (_Float16)src[i];
}

// ---------------------------------------------------------------------------
// Generic C = A @ B^T + bias, f16 in / f16 out, double-buffered LDS pipeline.
// Block tile 64(M) x 64(N): 4 waves stacked on M, B tile shared via LDS.
//   mode 1: head-split store  C[((b*16+h)*S + s)*64 + d]
//   mode 2: head-split transposed store C[((b*16+h)*64 + d)*S + s]
// ---------------------------------------------------------------------------
__global__ __launch_bounds__(128)
void gemm_bt_proj_kernel(const _Float16* __restrict__ A,
                         const _Float16* __restrict__ Bw,
                         const float* __restrict__ bias,
                         _Float16* __restrict__ C,
                         int N, int K, int S, int mode) {
  __shared__ __align__(16) _Float16 sA[2][64 * LDH];
  __shared__ __align__(16) _Float16 sB[2][64 * LDH];

  const int wid = threadIdx.x >> 5;
  const int m0  = blockIdx.y * 64 + wid * 16;
  const int n0  = blockIdx.x * 64;
  const _Float16* Ag = A  + (size_t)(blockIdx.y * 64) * K;
  const _Float16* Bg = Bw + (size_t)n0 * K;

  v8f acc[4];
#pragma unroll
  for (int j = 0; j < 4; ++j) acc[j] = (v8f){};

  const int nst = K / 32;
  stage_f16f16(Ag, K, Bg, K, 0, sA[0], sB[0]);
  for (int s = 0; s < nst; ++s) {
    const int cur = s & 1;
    if (s + 1 < nst) {
      stage_f16f16(Ag, K, Bg, K, (s + 1) * 32, sA[cur ^ 1], sB[cur ^ 1]);
      WAIT_ASYNC(4);
    } else {
      WAIT_ASYNC(0);
    }
    __syncthreads();
    v16h a = frag_lds_f16(sA[cur] + wid * 16 * LDH);
#pragma unroll
    for (int j = 0; j < 4; ++j) {
      v16h b = frag_lds_f16(sB[cur] + j * 16 * LDH);
      acc[j] = __builtin_amdgcn_wmma_f32_16x16x32_f16(
          false, a, false, b, (short)0, acc[j], false, false);
    }
    __syncthreads();
  }

  const int lane = threadIdx.x & 31;
  const int col  = lane & 15;
  const int rb   = (lane >> 4) * 8;
#pragma unroll
  for (int j = 0; j < 4; ++j) {
    const int n = n0 + 16 * j + col;
    const float bv = bias ? bias[n] : 0.0f;
#pragma unroll
    for (int i = 0; i < 8; ++i) {
      const int m = m0 + rb + i;
      const float val = acc[j][i] + bv;
      const int bb = m / S, ss = m % S;
      const int h = n >> 6, d = n & 63;
      size_t idx;
      if (mode == 1) idx = ((size_t)(bb * NH + h) * S + ss) * DH + d;
      else           idx = ((size_t)(bb * NH + h) * DH + d) * (size_t)S + ss;
      C[idx] = (_Float16)val;
    }
  }
}

// ---------------------------------------------------------------------------
// scores = (q @ k^T) / 8, masked -> f32 att_weights region of d_out.
//   qh: [B,H,SQ,64], kh: [B,H,SKV,64], mask: [B,SKV,SQ] bytes (true -> -1e9)
// ---------------------------------------------------------------------------
__global__ __launch_bounds__(128)
void attn_scores_kernel(const _Float16* __restrict__ qh,
                        const _Float16* __restrict__ kh,
                        const unsigned char* __restrict__ mask,
                        float* __restrict__ w) {
  __shared__ __align__(16) _Float16 sA[2][64 * LDH];
  __shared__ __align__(16) _Float16 sB[2][64 * LDH];

  const int bh  = blockIdx.z;
  const int b   = bh >> 4;
  const int wid = threadIdx.x >> 5;
  const int m0  = blockIdx.y * 64 + wid * 16;
  const int n0  = blockIdx.x * 64;

  const _Float16* Ag = qh + (size_t)bh * SQ * DH + (size_t)(blockIdx.y * 64) * DH;
  const _Float16* Bg = kh + (size_t)bh * SKV * DH + (size_t)n0 * DH;

  v8f acc[4];
#pragma unroll
  for (int j = 0; j < 4; ++j) acc[j] = (v8f){};

  stage_f16f16(Ag, DH, Bg, DH, 0, sA[0], sB[0]);
#pragma unroll
  for (int s = 0; s < 2; ++s) {               // K = DH = 64 -> 2 stages
    const int cur = s & 1;
    if (s == 0) {
      stage_f16f16(Ag, DH, Bg, DH, 32, sA[1], sB[1]);
      WAIT_ASYNC(4);
    } else {
      WAIT_ASYNC(0);
    }
    __syncthreads();
    v16h a = frag_lds_f16(sA[cur] + wid * 16 * LDH);
#pragma unroll
    for (int j = 0; j < 4; ++j) {
      v16h bf = frag_lds_f16(sB[cur] + j * 16 * LDH);
      acc[j] = __builtin_amdgcn_wmma_f32_16x16x32_f16(
          false, a, false, bf, (short)0, acc[j], false, false);
    }
    __syncthreads();
  }

  const int lane = threadIdx.x & 31;
  const int col  = lane & 15;
  const int rb   = (lane >> 4) * 8;
  float* wrow = w + (size_t)bh * SQ * SKV;
#pragma unroll
  for (int j = 0; j < 4; ++j) {
    const int n = n0 + 16 * j + col;
    const unsigned char* mp = mask + ((size_t)b * SKV + n) * SQ + m0 + rb;
#pragma unroll
    for (int i = 0; i < 8; ++i) {
      const int m = m0 + rb + i;
      float val = acc[j][i] * 0.125f;         // 1/sqrt(DH=64)
      if (mp[i]) val = -1e9f;
      wrow[(size_t)m * SKV + n] = val;
    }
  }
}

// ---------------------------------------------------------------------------
// In-place row softmax over last dim (SKV=4096). One 256-thread block / row.
// ---------------------------------------------------------------------------
__global__ __launch_bounds__(256)
void softmax_kernel(float* __restrict__ w) {
  float* p = w + (size_t)blockIdx.x * SKV;
  const int tid  = threadIdx.x;
  const int lane = tid & 31;
  const int wid  = tid >> 5;
  __shared__ float red[8];

  float v[16];
  float lmax = -3.4e38f;
#pragma unroll
  for (int i = 0; i < 16; ++i) {
    v[i] = p[tid + 256 * i];
    lmax = fmaxf(lmax, v[i]);
  }
#pragma unroll
  for (int off = 16; off > 0; off >>= 1)
    lmax = fmaxf(lmax, __shfl_xor(lmax, off, 32));
  if (lane == 0) red[wid] = lmax;
  __syncthreads();
  if (tid == 0) {
    float m = red[0];
#pragma unroll
    for (int i = 1; i < 8; ++i) m = fmaxf(m, red[i]);
    red[0] = m;
  }
  __syncthreads();
  const float rmax = red[0];
  __syncthreads();

  float lsum = 0.0f;
#pragma unroll
  for (int i = 0; i < 16; ++i) {
    v[i] = __expf(v[i] - rmax);
    lsum += v[i];
  }
#pragma unroll
  for (int off = 16; off > 0; off >>= 1)
    lsum += __shfl_xor(lsum, off, 32);
  if (lane == 0) red[wid] = lsum;
  __syncthreads();
  if (tid == 0) {
    float s = red[0];
#pragma unroll
    for (int i = 1; i < 8; ++i) s += red[i];
    red[0] = s;
  }
  __syncthreads();
  const float inv = 1.0f / red[0];

#pragma unroll
  for (int i = 0; i < 16; ++i) p[tid + 256 * i] = v[i] * inv;
}

// ---------------------------------------------------------------------------
// out_head = att_weights @ v  (K = SKV = 4096 -> 128 WMMA steps).
//   w:  [B,H,SQ,SKV] f32 -> staged in LDS, converted to f16 at fragment build
//   vT: [B,H,64,SKV] f16
//   attout: [B,SQ,H*64] f16
// ---------------------------------------------------------------------------
__global__ __launch_bounds__(128)
void attn_av_kernel(const float* __restrict__ w,
                    const _Float16* __restrict__ vT,
                    _Float16* __restrict__ attout) {
  __shared__ __align__(16) float    sA[2][64 * LDF];
  __shared__ __align__(16) _Float16 sB[2][64 * LDH];

  const int bh  = blockIdx.z;
  const int b   = bh >> 4;
  const int h   = bh & 15;
  const int wid = threadIdx.x >> 5;
  const int m0  = blockIdx.y * 64 + wid * 16;

  const float*    Ag = w  + (size_t)bh * SQ * SKV + (size_t)(blockIdx.y * 64) * SKV;
  const _Float16* Bg = vT + (size_t)bh * DH * SKV;

  v8f acc[4];
#pragma unroll
  for (int j = 0; j < 4; ++j) acc[j] = (v8f){};

  const int nst = SKV / 32;   // 128
  stage_f32f16(Ag, SKV, Bg, SKV, 0, sA[0], sB[0]);
  for (int s = 0; s < nst; ++s) {
    const int cur = s & 1;
    if (s + 1 < nst) {
      stage_f32f16(Ag, SKV, Bg, SKV, (s + 1) * 32, sA[cur ^ 1], sB[cur ^ 1]);
      WAIT_ASYNC(6);
    } else {
      WAIT_ASYNC(0);
    }
    __syncthreads();
    v16h a = frag_lds_f32(sA[cur] + wid * 16 * LDF);
#pragma unroll
    for (int j = 0; j < 4; ++j) {
      v16h bf = frag_lds_f16(sB[cur] + j * 16 * LDH);
      acc[j] = __builtin_amdgcn_wmma_f32_16x16x32_f16(
          false, a, false, bf, (short)0, acc[j], false, false);
    }
    __syncthreads();
  }

  const int lane = threadIdx.x & 31;
  const int col  = lane & 15;
  const int rb   = (lane >> 4) * 8;
#pragma unroll
  for (int j = 0; j < 4; ++j) {
    const int d = 16 * j + col;
#pragma unroll
    for (int i = 0; i < 8; ++i) {
      const int m = m0 + rb + i;
      attout[((size_t)b * SQ + m) * (NH * DH) + h * DH + d] = (_Float16)acc[j][i];
    }
  }
}

// ---------------------------------------------------------------------------
// Final projection: att_output = attout @ Wo^T + bo -> f32 front of d_out.
// ---------------------------------------------------------------------------
__global__ __launch_bounds__(128)
void gemm_bt_out_kernel(const _Float16* __restrict__ A,
                        const _Float16* __restrict__ Bw,
                        const float* __restrict__ bias,
                        float* __restrict__ C, int N, int K) {
  __shared__ __align__(16) _Float16 sA[2][64 * LDH];
  __shared__ __align__(16) _Float16 sB[2][64 * LDH];

  const int wid = threadIdx.x >> 5;
  const int m0  = blockIdx.y * 64 + wid * 16;
  const int n0  = blockIdx.x * 64;
  const _Float16* Ag = A  + (size_t)(blockIdx.y * 64) * K;
  const _Float16* Bg = Bw + (size_t)n0 * K;

  v8f acc[4];
#pragma unroll
  for (int j = 0; j < 4; ++j) acc[j] = (v8f){};

  const int nst = K / 32;
  stage_f16f16(Ag, K, Bg, K, 0, sA[0], sB[0]);
  for (int s = 0; s < nst; ++s) {
    const int cur = s & 1;
    if (s + 1 < nst) {
      stage_f16f16(Ag, K, Bg, K, (s + 1) * 32, sA[cur ^ 1], sB[cur ^ 1]);
      WAIT_ASYNC(4);
    } else {
      WAIT_ASYNC(0);
    }
    __syncthreads();
    v16h a = frag_lds_f16(sA[cur] + wid * 16 * LDH);
#pragma unroll
    for (int j = 0; j < 4; ++j) {
      v16h bf = frag_lds_f16(sB[cur] + j * 16 * LDH);
      acc[j] = __builtin_amdgcn_wmma_f32_16x16x32_f16(
          false, a, false, bf, (short)0, acc[j], false, false);
    }
    __syncthreads();
  }

  const int lane = threadIdx.x & 31;
  const int col  = lane & 15;
  const int rb   = (lane >> 4) * 8;
#pragma unroll
  for (int j = 0; j < 4; ++j) {
    const int n = n0 + 16 * j + col;
    const float bv = bias[n];
#pragma unroll
    for (int i = 0; i < 8; ++i) {
      const int m = m0 + rb + i;
      C[(size_t)m * N + n] = acc[j][i] + bv;
    }
  }
}

// ---------------------------------------------------------------------------
// Host launcher
// ---------------------------------------------------------------------------
static inline size_t align256(size_t x) { return (x + 255) & ~(size_t)255; }

extern "C" void kernel_launch(void* const* d_in, const int* in_sizes, int n_in,
                              void* d_out, int out_size, void* d_ws, size_t ws_size,
                              hipStream_t stream) {
  (void)in_sizes; (void)n_in; (void)out_size; (void)ws_size;

  const float* Q    = (const float*)d_in[0];
  const float* K    = (const float*)d_in[1];
  const float* V    = (const float*)d_in[2];
  const unsigned char* mask = (const unsigned char*)d_in[3];
  const float* Wq   = (const float*)d_in[4];
  const float* bq   = (const float*)d_in[5];
  const float* Wk   = (const float*)d_in[6];
  const float* bk   = (const float*)d_in[7];
  const float* Wv   = (const float*)d_in[8];
  const float* bv   = (const float*)d_in[9];
  const float* Wo   = (const float*)d_in[10];
  const float* bo   = (const float*)d_in[11];

  float* out_att = (float*)d_out;                                   // [B,SQ,256]
  float* out_w   = (float*)d_out + (size_t)BB * SQ * DIMOUT;        // [B,H,SQ,SKV]

  // workspace carve (all f16)
  char* ws = (char*)d_ws;
  size_t off = 0;
  auto carve = [&](size_t elems) {
    _Float16* p = (_Float16*)(ws + off);
    off = align256(off + elems * sizeof(_Float16));
    return p;
  };
  _Float16* Qh   = carve((size_t)BB * SQ * DIMQ);
  _Float16* Kh   = carve((size_t)BB * SKV * DIMKV);
  _Float16* Vh   = carve((size_t)BB * SKV * DIMKV);
  _Float16* Wqh  = carve((size_t)DIMQ * DIMQ);
  _Float16* Wkh  = carve((size_t)DIMQ * DIMKV);
  _Float16* Wvh  = carve((size_t)DIMQ * DIMKV);
  _Float16* Woh  = carve((size_t)DIMOUT * DIMQ);
  _Float16* qh   = carve((size_t)BB * NH * SQ * DH);
  _Float16* kh   = carve((size_t)BB * NH * SKV * DH);
  _Float16* vTh  = carve((size_t)BB * NH * DH * SKV);
  _Float16* aoh  = carve((size_t)BB * SQ * NH * DH);

  // 1) f32 -> f16 conversions
  auto cvt = [&](const float* s, _Float16* d, long n) {
    int blocks = (int)((n + 1023) / 1024);
    if (blocks > 4096) blocks = 4096;
    cvt_f32_f16_kernel<<<blocks, 256, 0, stream>>>(s, d, n);
  };
  cvt(Q,  Qh,  (long)BB * SQ * DIMQ);
  cvt(K,  Kh,  (long)BB * SKV * DIMKV);
  cvt(V,  Vh,  (long)BB * SKV * DIMKV);
  cvt(Wq, Wqh, (long)DIMQ * DIMQ);
  cvt(Wk, Wkh, (long)DIMQ * DIMKV);
  cvt(Wv, Wvh, (long)DIMQ * DIMKV);
  cvt(Wo, Woh, (long)DIMOUT * DIMQ);

  // 2) projections (WMMA): q -> head-split, k -> head-split, v -> transposed
  gemm_bt_proj_kernel<<<dim3(DIMQ / 64, (BB * SQ) / 64), 128, 0, stream>>>(
      Qh, Wqh, bq, qh, DIMQ, DIMQ, SQ, 1);
  gemm_bt_proj_kernel<<<dim3(DIMQ / 64, (BB * SKV) / 64), 128, 0, stream>>>(
      Kh, Wkh, bk, kh, DIMQ, DIMKV, SKV, 1);
  gemm_bt_proj_kernel<<<dim3(DIMQ / 64, (BB * SKV) / 64), 128, 0, stream>>>(
      Vh, Wvh, bv, vTh, DIMQ, DIMKV, SKV, 2);

  // 3) masked, scaled scores -> d_out weights region (WMMA)
  attn_scores_kernel<<<dim3(SKV / 64, SQ / 64, BB * NH), 128, 0, stream>>>(
      qh, kh, mask, out_w);

  // 4) row softmax in place
  softmax_kernel<<<BB * NH * SQ, 256, 0, stream>>>(out_w);

  // 5) weights @ v (WMMA, LDS-staged f32 A)
  attn_av_kernel<<<dim3(1, SQ / 64, BB * NH), 128, 0, stream>>>(out_w, vTh, aoh);

  // 6) output projection (WMMA) -> f32 att_output
  gemm_bt_out_kernel<<<dim3(DIMOUT / 64, (BB * SQ) / 64), 128, 0, stream>>>(
      aoh, Woh, bo, out_att, DIMOUT, DIMQ);
}
